// Block_17386027614858
// MI455X (gfx1250) — matrix operand order repaired and checked by reading, not compiled, not run on hardware
//
#include <hip/hip_runtime.h>
#include <cstdint>
#include <cstddef>

// ---------------------------------------------------------------------------
// CDNA5 (gfx1250) DiT block: AdaLN-RMSNorm -> GQA attn (RoPE, causal, flash)
// -> gated residual -> AdaLN-RMSNorm -> gated-GeLU FFN -> gated residual.
// All matmuls in bf16 WMMA (v_wmma_f32_16x16x32_bf16) with fp32 accumulation.
// GEMMs are LDS double-buffered (ping-pong, one barrier per 32-K stage).
// ---------------------------------------------------------------------------

typedef __attribute__((ext_vector_type(16))) __bf16 v16bf;
typedef __attribute__((ext_vector_type(8)))  float  v8f;

union FragBF { v16bf v; unsigned short u[16]; };
union FragF  { v8f   v; float          f[8];  };
union U16x8  { uint4 q; unsigned short s[8];  };

__device__ __forceinline__ unsigned short f2bf(float f) {
  unsigned int u = __float_as_uint(f);
  u += 0x7FFFu + ((u >> 16) & 1u);          // round-to-nearest-even
  return (unsigned short)(u >> 16);
}
__device__ __forceinline__ float bf2f(unsigned short h) {
  return __uint_as_float(((unsigned int)h) << 16);
}

// ---------------------------------------------------------------------------
// Weight conversion kernels
// ---------------------------------------------------------------------------
__global__ void cvt_f32_bf16_kernel(const float* __restrict__ src,
                                    unsigned short* __restrict__ dst,
                                    long long nelem) {
  long long i = (long long)blockIdx.x * blockDim.x + threadIdx.x;
  long long stride = (long long)gridDim.x * blockDim.x;
  for (; i < nelem; i += stride) dst[i] = f2bf(src[i]);
}

// Wq [N,D,H] -> dst [D][N*H]
__global__ void cvt_wq_kernel(const float* __restrict__ Wq,
                              unsigned short* __restrict__ dst,
                              int NH, int D, int H) {
  long long total = (long long)NH * D * H;
  long long idx = (long long)blockIdx.x * blockDim.x + threadIdx.x;
  long long stride = (long long)gridDim.x * blockDim.x;
  for (; idx < total; idx += stride) {
    int n = (int)(idx / ((long long)D * H));
    long long rem = idx % ((long long)D * H);
    int d = (int)(rem / H);
    int h = (int)(rem % H);
    dst[(size_t)d * NH * H + (size_t)n * H + h] = f2bf(Wq[idx]);
  }
}

// ---------------------------------------------------------------------------
// AdaLN modulation: mod[b][c] = cond[b] . Wmod[:,c] + bias[c]   (B = 2)
// Coalesced over columns; both batch rows accumulated in one sweep of W.
// ---------------------------------------------------------------------------
__global__ void modproj_kernel(const float* __restrict__ cond,
                               const float* __restrict__ W,
                               const float* __restrict__ bias,
                               float* __restrict__ mod,
                               int D, int threeD) {
  int col = blockIdx.x * blockDim.x + threadIdx.x;
  if (col >= threeD) return;
  float s0 = bias[col], s1 = bias[col];
  for (int d = 0; d < D; ++d) {
    float w = W[(size_t)d * threeD + col];
    s0 = fmaf(cond[d], w, s0);
    s1 = fmaf(cond[D + d], w, s1);
  }
  mod[col] = s0;
  mod[(size_t)threeD + col] = s1;
}

// ---------------------------------------------------------------------------
// RMSNorm (fp32) + AdaLN scale/shift -> bf16. One block per row.
// ---------------------------------------------------------------------------
__global__ __launch_bounds__(256) void adaln_kernel(
    const float* __restrict__ x, const float* __restrict__ mod,
    unsigned short* __restrict__ out, int D, int T, int threeD) {
  const int row = blockIdx.x;
  const int b = row / T;
  const int tid = threadIdx.x;
  const float* xr = x + (size_t)row * D;
  float ss = 0.f;
  for (int c = tid; c < D; c += 256) { float v = xr[c]; ss = fmaf(v, v, ss); }
#pragma unroll
  for (int msk = 1; msk < 32; msk <<= 1) ss += __shfl_xor(ss, msk, 32);
  __shared__ float red[8];
  if ((tid & 31) == 0) red[tid >> 5] = ss;
  __syncthreads();
  float tot = 0.f;
#pragma unroll
  for (int w = 0; w < 8; ++w) tot += red[w];
  const float rinv = rsqrtf(tot / (float)D + 1e-6f);
  const float* mrow = mod + (size_t)b * threeD;
  for (int c = tid; c < D; c += 256) {
    float sc = mrow[c], sh = mrow[D + c];
    out[(size_t)row * D + c] = f2bf(xr[c] * rinv * (1.f + sc) + sh);
  }
}

// ---------------------------------------------------------------------------
// RoPE + head-major relayout:  src [b][t][NH][H] -> dst [b][NH][t][H]
// out = rope(x) * outscale   (outscale = H^-0.5 for q, 1 for k)
// ---------------------------------------------------------------------------
__global__ void rope_kernel(const unsigned short* __restrict__ src,
                            const float* __restrict__ positions,
                            unsigned short* __restrict__ dst,
                            int B, int T, int NH, int H, float outscale) {
  const int half = H / 2;
  long long total = (long long)B * T * NH * half;
  long long idx = (long long)blockIdx.x * blockDim.x + threadIdx.x;
  if (idx >= total) return;
  int i = (int)(idx % half);
  long long t1 = idx / half;
  int n = (int)(t1 % NH);
  long long t2 = t1 / NH;
  int t = (int)(t2 % T);
  int b = (int)(t2 / T);
  float p = positions[(size_t)b * T + t];
  float fe = (2.0f / (float)H) * (float)i;
  float ts = __powf(10000.0f, fe);
  float rad = p / ts;
  float sn, cs;
  __sincosf(rad, &sn, &cs);
  const unsigned short* sp = src + (((size_t)b * T + t) * NH + n) * H;
  float x1 = bf2f(sp[i]), x2 = bf2f(sp[i + half]);
  unsigned short* dp = dst + (((size_t)b * NH + n) * T + t) * H;
  dp[i]        = f2bf((x1 * cs - x2 * sn) * outscale);
  dp[i + half] = f2bf((x2 * cs + x1 * sn) * outscale);
}

// ---------------------------------------------------------------------------
// Generic bf16 WMMA GEMM: C[M,N] = A[M,K] * B[K,N]   (all row-major)
// Block tile 64x128, K-step 32, 8 waves (2x4), wave tile 32x32.
// LDS ping-pong double buffering: next stage's global_load_b128s are issued
// into registers before the WMMA stage, committed to the alternate buffer
// after it; one s_barrier per stage.
// B is staged N-major ("transposed") in LDS so B-fragments are 16 contiguous
// K-elements per lane (k = 16*hi + e; A fragments use the same order).
// Epilogues: 0 = store bf16
//            1 = fp32 store: xres + acc*gate[b][col]   (gated residual)
//            2 = dual-B: bf16 store gelu(acc0)*acc1    (fused FFN gate)
// ---------------------------------------------------------------------------
constexpr int BM = 64, BN = 128, BK = 32;

template <int EPI>
__global__ __launch_bounds__(256) void gemm_bf16_kernel(
    const unsigned short* __restrict__ A,
    const unsigned short* __restrict__ B0,
    const unsigned short* __restrict__ B1,
    void* __restrict__ Cout,
    const float* __restrict__ xres,
    const float* __restrict__ gate, int gate_stride,
    int M, int N, int Kd, int T) {
  constexpr bool DUAL = (EPI == 2);
  __shared__ unsigned short As[2][BM][BK + 8];
  __shared__ unsigned short Bs0[2][BN][BK + 8];
  __shared__ unsigned short Bs1[DUAL ? 2 : 1][BN][BK + 8];

  const int tid = threadIdx.x;
  const int lane = tid & 31;
  const int wave = tid >> 5;
  const int wm = wave & 1, wn = wave >> 1;
  const int blockM = blockIdx.y * BM;
  const int blockN = blockIdx.x * BN;
  const int mlow = lane & 15;
  const int hi = (lane & 16) ? 1 : 0;

  FragF acc0[2][2], acc1[2][2];
#pragma unroll
  for (int i = 0; i < 2; ++i)
#pragma unroll
    for (int j = 0; j < 2; ++j)
#pragma unroll
      for (int r = 0; r < 8; ++r) {
        acc0[i][j].f[r] = 0.f;
        if (DUAL) acc1[i][j].f[r] = 0.f;
      }

  const int arow = tid >> 2, acol = (tid & 3) * 8;     // 64x32 A tile
  const int brow = tid >> 3, bcol = (tid & 7) * 16;    // 32x128 B tile
  const unsigned short* Aptr  = A + (size_t)(blockM + arow) * Kd + acol;
  const unsigned short* B0ptr = B0 + (size_t)brow * N + blockN + bcol;
  const unsigned short* B1ptr = DUAL ? (B1 + (size_t)brow * N + blockN + bcol) : nullptr;

  // prologue: fill buffer 0 with stage 0
  {
    U16x8 ta, t0, t1;
    ta.q = *reinterpret_cast<const uint4*>(Aptr);
    t0.q = *reinterpret_cast<const uint4*>(B0ptr);
    t1.q = *reinterpret_cast<const uint4*>(B0ptr + 8);
#pragma unroll
    for (int e = 0; e < 8; ++e) {
      As[0][arow][acol + e] = ta.s[e];
      Bs0[0][bcol + e][brow] = t0.s[e];
      Bs0[0][bcol + 8 + e][brow] = t1.s[e];
    }
    if (DUAL) {
      t0.q = *reinterpret_cast<const uint4*>(B1ptr);
      t1.q = *reinterpret_cast<const uint4*>(B1ptr + 8);
#pragma unroll
      for (int e = 0; e < 8; ++e) {
        Bs1[0][bcol + e][brow] = t0.s[e];
        Bs1[0][bcol + 8 + e][brow] = t1.s[e];
      }
    }
  }
  __syncthreads();

  const int nstages = Kd / BK;
  for (int s = 0; s < nstages; ++s) {
    const int cur = s & 1, nxt = cur ^ 1;
    const bool have_next = (s + 1 < nstages);

    // issue next stage's global loads early (latency overlapped with WMMA)
    U16x8 pa, pb0a, pb0b, pb1a, pb1b;
    if (have_next) {
      const int k1 = (s + 1) * BK;
      pa.q   = *reinterpret_cast<const uint4*>(Aptr + k1);
      pb0a.q = *reinterpret_cast<const uint4*>(B0ptr + (size_t)k1 * N);
      pb0b.q = *reinterpret_cast<const uint4*>(B0ptr + (size_t)k1 * N + 8);
      if (DUAL) {
        pb1a.q = *reinterpret_cast<const uint4*>(B1ptr + (size_t)k1 * N);
        pb1b.q = *reinterpret_cast<const uint4*>(B1ptr + (size_t)k1 * N + 8);
      }
    }

    // WMMA stage from buffer `cur`
    FragBF af[2], bf0[2], bf1[2];
#pragma unroll
    for (int i = 0; i < 2; ++i) {  // A frag: two 8-elem K runs per lane
      int m = wm * 32 + i * 16 + mlow;
      int kb = hi ? 16 : 0;
#pragma unroll
      for (int e = 0; e < 8; ++e) {
        af[i].u[e] = As[cur][m][kb + e];
        af[i].u[8 + e] = As[cur][m][kb + 8 + e];
      }
    }
#pragma unroll
    for (int j = 0; j < 2; ++j) {  // B frag: 16 contiguous K per lane
      int nn = wn * 32 + j * 16 + mlow;
      int kb = hi ? 16 : 0;
#pragma unroll
      for (int e = 0; e < 16; ++e) bf0[j].u[e] = Bs0[cur][nn][kb + e];
      if (DUAL) {
#pragma unroll
        for (int e = 0; e < 16; ++e) bf1[j].u[e] = Bs1[cur][nn][kb + e];
      }
    }
#pragma unroll
    for (int i = 0; i < 2; ++i)
#pragma unroll
      for (int j = 0; j < 2; ++j) {
        acc0[i][j].v = __builtin_amdgcn_wmma_f32_16x16x32_bf16(
            false, af[i].v, false, bf0[j].v, (short)0, acc0[i][j].v, false, false);
        if (DUAL)
          acc1[i][j].v = __builtin_amdgcn_wmma_f32_16x16x32_bf16(
              false, af[i].v, false, bf1[j].v, (short)0, acc1[i][j].v, false, false);
      }

    // commit prefetch into buffer `nxt`
    if (have_next) {
#pragma unroll
      for (int e = 0; e < 8; ++e) {
        As[nxt][arow][acol + e] = pa.s[e];
        Bs0[nxt][bcol + e][brow] = pb0a.s[e];
        Bs0[nxt][bcol + 8 + e][brow] = pb0b.s[e];
      }
      if (DUAL) {
#pragma unroll
        for (int e = 0; e < 8; ++e) {
          Bs1[nxt][bcol + e][brow] = pb1a.s[e];
          Bs1[nxt][bcol + 8 + e][brow] = pb1b.s[e];
        }
      }
    }
    __syncthreads();
  }

  // Epilogue. C layout: lane = column (mod 16), VGPR r = row r + 8*hi.
#pragma unroll
  for (int i = 0; i < 2; ++i) {
#pragma unroll
    for (int j = 0; j < 2; ++j) {
      const int col = blockN + wn * 32 + j * 16 + mlow;
#pragma unroll
      for (int r = 0; r < 8; ++r) {
        const int row = blockM + wm * 32 + i * 16 + hi * 8 + r;
        if (EPI == 0) {
          ((unsigned short*)Cout)[(size_t)row * N + col] = f2bf(acc0[i][j].f[r]);
        } else if (EPI == 1) {
          const int bidx = row / T;
          const float g = gate[(size_t)bidx * gate_stride + col];
          ((float*)Cout)[(size_t)row * N + col] =
              xres[(size_t)row * N + col] + acc0[i][j].f[r] * g;
        } else {
          const float a0 = acc0[i][j].f[r], a1 = acc1[i][j].f[r];
          const float t = tanhf(0.7978845608028654f * (a0 + 0.044715f * a0 * a0 * a0));
          ((unsigned short*)Cout)[(size_t)row * N + col] = f2bf(0.5f * a0 * (1.f + t) * a1);
        }
      }
    }
  }
}

// ---------------------------------------------------------------------------
// Flash attention (causal, 1 kv head). Block = 8 waves handling 128 q-rows of
// one (batch, q-head); K tile (row-major == B^T layout for QK^T since the
// contraction is over H) and transposed V tile shared in LDS. Online softmax
// in WMMA C-layout; P re-staged via per-wave LDS into A-fragment layout.
// Single-buffered: the long softmax VALU phase + 8-wave occupancy hide the
// K/V load latency; ~7% of total FLOPs.
// ---------------------------------------------------------------------------
__global__ __launch_bounds__(256) void flash_attn_kernel(
    const unsigned short* __restrict__ qr,   // [B][NH][T][H], pre-scaled H^-0.5
    const unsigned short* __restrict__ kr,   // [B][T][H]
    const unsigned short* __restrict__ vr,   // [B][T][H]
    unsigned short* __restrict__ enc,        // [B*T][NH*H]
    int T, int NH, int H /* == 256 */) {
  __shared__ unsigned short Ks[32][264];     // 32 keys x H (padded)
  __shared__ unsigned short Vt[256][40];     // H x 32 keys (padded)
  __shared__ unsigned short Ps[8][16][34];   // per-wave P staging

  const int b = blockIdx.z, n = blockIdx.y;
  const int qbase = blockIdx.x * 128;
  const int tid = threadIdx.x, lane = tid & 31, wave = tid >> 5;
  const int qrow = qbase + wave * 16;
  const int mlow = lane & 15;
  const int hi = (lane & 16) ? 1 : 0;

  // Q fragments (16 rows x 256, 8 K-chunks of 32) held in registers
  const unsigned short* Q = qr + (((size_t)b * NH + n) * T + qrow) * H;
  FragBF qf[8];
#pragma unroll
  for (int j = 0; j < 8; ++j) {
    const unsigned short* p = Q + (size_t)mlow * H + j * 32 + (hi ? 16 : 0);
#pragma unroll
    for (int e = 0; e < 8; ++e) { qf[j].u[e] = p[e]; qf[j].u[8 + e] = p[8 + e]; }
  }

  FragF o[16];
#pragma unroll
  for (int c = 0; c < 16; ++c)
#pragma unroll
    for (int r = 0; r < 8; ++r) o[c].f[r] = 0.f;
  float mrow[8], lrow[8];
#pragma unroll
  for (int r = 0; r < 8; ++r) { mrow[r] = -3.0e38f; lrow[r] = 0.f; }

  const int srow_ld = tid >> 3, hb_ld = (tid & 7) * 32;
  const int smax = qbase + 128;
  for (int s0 = 0; s0 < smax; s0 += 32) {
    __syncthreads();
    {  // cooperative K (row-major) and V (transposed) tile loads
      const unsigned short* ks = kr + ((size_t)b * T + s0 + srow_ld) * H + hb_ld;
      const unsigned short* vs = vr + ((size_t)b * T + s0 + srow_ld) * H + hb_ld;
#pragma unroll
      for (int q4 = 0; q4 < 4; ++q4) {
        U16x8 tk, tv;
        tk.q = *reinterpret_cast<const uint4*>(ks + q4 * 8);
        tv.q = *reinterpret_cast<const uint4*>(vs + q4 * 8);
#pragma unroll
        for (int e = 0; e < 8; ++e) {
          Ks[srow_ld][hb_ld + q4 * 8 + e] = tk.s[e];
          Vt[hb_ld + q4 * 8 + e][srow_ld] = tv.s[e];
        }
      }
    }
    __syncthreads();

    // logits: 16 rows x 32 keys = two 16x16 C tiles, contraction over H
    FragF lg[2];
#pragma unroll
    for (int c = 0; c < 2; ++c)
#pragma unroll
      for (int r = 0; r < 8; ++r) lg[c].f[r] = 0.f;
#pragma unroll
    for (int j = 0; j < 8; ++j) {
#pragma unroll
      for (int c = 0; c < 2; ++c) {
        FragBF kf;
        const int nn = c * 16 + mlow;
        const int kb = j * 32 + (hi ? 16 : 0);
#pragma unroll
        for (int e = 0; e < 16; ++e) kf.u[e] = Ks[nn][kb + e];
        lg[c].v = __builtin_amdgcn_wmma_f32_16x16x32_bf16(
            false, qf[j].v, false, kf.v, (short)0, lg[c].v, false, false);
      }
    }

    // causal mask + online softmax (row stats via 16-lane xor reductions)
#pragma unroll
    for (int r = 0; r < 8; ++r) {
      const int row = qrow + hi * 8 + r;
#pragma unroll
      for (int c = 0; c < 2; ++c) {
        const int col = s0 + c * 16 + mlow;
        if (col > row) lg[c].f[r] = -2.3819763e38f;
      }
      float pm = fmaxf(lg[0].f[r], lg[1].f[r]);
#pragma unroll
      for (int msk = 1; msk < 16; msk <<= 1) pm = fmaxf(pm, __shfl_xor(pm, msk, 32));
      const float mnew = fmaxf(mrow[r], pm);
      const float corr = __expf(mrow[r] - mnew);
      mrow[r] = mnew;
      float rs = 0.f;
#pragma unroll
      for (int c = 0; c < 2; ++c) {
        float pv = __expf(lg[c].f[r] - mnew);
        lg[c].f[r] = pv;
        rs += pv;
      }
#pragma unroll
      for (int msk = 1; msk < 16; msk <<= 1) rs += __shfl_xor(rs, msk, 32);
      lrow[r] = lrow[r] * corr + rs;
#pragma unroll
      for (int cc = 0; cc < 16; ++cc) o[cc].f[r] *= corr;
    }

    // stage P (C layout -> LDS -> A-fragment layout); per-wave region
#pragma unroll
    for (int c = 0; c < 2; ++c)
#pragma unroll
      for (int r = 0; r < 8; ++r)
        Ps[wave][hi * 8 + r][c * 16 + mlow] = f2bf(lg[c].f[r]);
    FragBF pf;
    {
      const int kb = hi ? 16 : 0;
#pragma unroll
      for (int e = 0; e < 8; ++e) {
        pf.u[e] = Ps[wave][mlow][kb + e];
        pf.u[8 + e] = Ps[wave][mlow][kb + 8 + e];
      }
    }
    // O += P * V  (contraction over 32 keys, 16 column tiles of H)
#pragma unroll
    for (int cc = 0; cc < 16; ++cc) {
      FragBF vf;
      const int nn = cc * 16 + mlow;
      const int kb = hi ? 16 : 0;
#pragma unroll
      for (int e = 0; e < 16; ++e) vf.u[e] = Vt[nn][kb + e];
      o[cc].v = __builtin_amdgcn_wmma_f32_16x16x32_bf16(
          false, pf.v, false, vf.v, (short)0, o[cc].v, false, false);
    }
  }

  // normalize and write enc[b*T+t][n*H+h] (bf16)
#pragma unroll
  for (int cc = 0; cc < 16; ++cc) {
    const int h = cc * 16 + mlow;
#pragma unroll
    for (int r = 0; r < 8; ++r) {
      const int row = qrow + hi * 8 + r;
      const float val = o[cc].f[r] / lrow[r];
      enc[((size_t)b * T + row) * ((size_t)NH * H) + (size_t)n * H + h] = f2bf(val);
    }
  }
}

// ---------------------------------------------------------------------------
// Host orchestration
// ---------------------------------------------------------------------------
extern "C" void kernel_launch(void* const* d_in, const int* in_sizes, int n_in,
                              void* d_out, int out_size, void* d_ws, size_t ws_size,
                              hipStream_t stream) {
  (void)in_sizes; (void)n_in; (void)out_size; (void)ws_size;
  constexpr int B = 2, T = 2048, D = 2048, F = 16384, NH = 8, H = 256;
  constexpr int M = B * T;
  constexpr int threeD = 3 * D;

  const float* x      = (const float*)d_in[0];
  const float* pos    = (const float*)d_in[1];
  /* d_in[2]: attn_mask = causal tril, applied analytically */
  const float* cond   = (const float*)d_in[3];
  const float* Wmod_a = (const float*)d_in[4];
  const float* bmod_a = (const float*)d_in[5];
  const float* Wq     = (const float*)d_in[6];
  const float* Wkv    = (const float*)d_in[7];
  const float* Wo     = (const float*)d_in[8];
  const float* Wmod_f = (const float*)d_in[9];
  const float* bmod_f = (const float*)d_in[10];
  const float* Wg     = (const float*)d_in[11];
  const float* Wl     = (const float*)d_in[12];
  float* out = (float*)d_out;

  char* p = (char*)d_ws;
  auto alloc = [&](size_t bytes) -> void* {
    void* r = (void*)p;
    p += (bytes + 255) & ~(size_t)255;
    return r;
  };
  unsigned short* wq_bf  = (unsigned short*)alloc((size_t)D * NH * H * 2);
  unsigned short* wk_bf  = (unsigned short*)alloc((size_t)D * H * 2);
  unsigned short* wv_bf  = (unsigned short*)alloc((size_t)D * H * 2);
  unsigned short* wo_bf  = (unsigned short*)alloc((size_t)NH * H * D * 2);
  unsigned short* wg0_bf = (unsigned short*)alloc((size_t)D * F * 2);
  unsigned short* wg1_bf = (unsigned short*)alloc((size_t)D * F * 2);
  unsigned short* wl_bf  = (unsigned short*)alloc((size_t)F * D * 2);
  float* mod_a = (float*)alloc((size_t)B * threeD * 4);
  float* mod_f = (float*)alloc((size_t)B * threeD * 4);
  unsigned short* n1  = (unsigned short*)alloc((size_t)M * D * 2);
  unsigned short* qb  = (unsigned short*)alloc((size_t)M * NH * H * 2);
  unsigned short* kb  = (unsigned short*)alloc((size_t)M * H * 2);
  unsigned short* vb  = (unsigned short*)alloc((size_t)M * H * 2);
  unsigned short* qr  = (unsigned short*)alloc((size_t)M * NH * H * 2);
  unsigned short* krr = (unsigned short*)alloc((size_t)M * H * 2);
  unsigned short* enc = (unsigned short*)alloc((size_t)M * NH * H * 2);
  float* x1 = (float*)alloc((size_t)M * D * 4);
  unsigned short* n2 = (unsigned short*)alloc((size_t)M * D * 2);
  unsigned short* hb = (unsigned short*)alloc((size_t)M * F * 2);

  // 1. one-time (per launch) fp32 -> bf16 weight conversion
  cvt_wq_kernel<<<4096, 256, 0, stream>>>(Wq, wq_bf, NH, D, H);
  cvt_f32_bf16_kernel<<<2048, 256, 0, stream>>>(Wkv, wk_bf, (long long)D * H);
  cvt_f32_bf16_kernel<<<2048, 256, 0, stream>>>(Wkv + (size_t)D * H, wv_bf, (long long)D * H);
  cvt_f32_bf16_kernel<<<4096, 256, 0, stream>>>(Wo, wo_bf, (long long)NH * H * D);
  cvt_f32_bf16_kernel<<<8192, 256, 0, stream>>>(Wg, wg0_bf, (long long)D * F);
  cvt_f32_bf16_kernel<<<8192, 256, 0, stream>>>(Wg + (size_t)D * F, wg1_bf, (long long)D * F);
  cvt_f32_bf16_kernel<<<8192, 256, 0, stream>>>(Wl, wl_bf, (long long)F * D);

  // 2. AdaLN modulation projections (fp32 GEMV)
  modproj_kernel<<<(threeD + 255) / 256, 256, 0, stream>>>(cond, Wmod_a, bmod_a, mod_a, D, threeD);
  modproj_kernel<<<(threeD + 255) / 256, 256, 0, stream>>>(cond, Wmod_f, bmod_f, mod_f, D, threeD);

  // 3. pre-attention AdaLN RMSNorm -> bf16
  adaln_kernel<<<M, 256, 0, stream>>>(x, mod_a, n1, D, T, threeD);

  // 4. QKV projections (WMMA)
  gemm_bf16_kernel<0><<<dim3((NH * H) / BN, M / BM), 256, 0, stream>>>(
      n1, wq_bf, nullptr, qb, nullptr, nullptr, 0, M, NH * H, D, T);
  gemm_bf16_kernel<0><<<dim3(H / BN, M / BM), 256, 0, stream>>>(
      n1, wk_bf, nullptr, kb, nullptr, nullptr, 0, M, H, D, T);
  gemm_bf16_kernel<0><<<dim3(H / BN, M / BM), 256, 0, stream>>>(
      n1, wv_bf, nullptr, vb, nullptr, nullptr, 0, M, H, D, T);

  // 5. RoPE + head-major relayout (q pre-scaled by H^-0.5 = 1/16)
  rope_kernel<<<(B * T * NH * (H / 2) + 255) / 256, 256, 0, stream>>>(
      qb, pos, qr, B, T, NH, H, 0.0625f);
  rope_kernel<<<(B * T * (H / 2) + 255) / 256, 256, 0, stream>>>(
      kb, pos, krr, B, T, 1, H, 1.0f);

  // 6. causal flash attention (WMMA)
  flash_attn_kernel<<<dim3(T / 128, NH, B), 256, 0, stream>>>(qr, krr, vb, enc, T, NH, H);

  // 7. output projection + gated residual: x1 = x + attn_out * gate_a
  gemm_bf16_kernel<1><<<dim3(D / BN, M / BM), 256, 0, stream>>>(
      enc, wo_bf, nullptr, x1, x, mod_a + 2 * D, threeD, M, D, NH * H, T);

  // 8. pre-FFN AdaLN RMSNorm
  adaln_kernel<<<M, 256, 0, stream>>>(x1, mod_f, n2, D, T, threeD);

  // 9. fused gated-GeLU FFN up: h = gelu(n2@Wg0) * (n2@Wg1)  (dual-B WMMA)
  gemm_bf16_kernel<2><<<dim3(F / BN, M / BM), 256, 0, stream>>>(
      n2, wg0_bf, wg1_bf, hb, nullptr, nullptr, 0, M, F, D, T);

  // 10. FFN down + gated residual -> out
  gemm_bf16_kernel<1><<<dim3(D / BN, M / BM), 256, 0, stream>>>(
      hb, wl_bf, nullptr, out, x1, mod_f + 2 * D, threeD, M, D, F, T);
}
